// Attention_37297495998691
// MI455X (gfx1250) — matrix operand order repaired and checked by reading, not compiled, not run on hardware
//
#include <hip/hip_runtime.h>
#include <hip/hip_bf16.h>

// ---------------------------------------------------------------------------
// Problem constants (from the reference)
// ---------------------------------------------------------------------------
#define NB    2
#define TSEQ  2048
#define DM    1536
#define NH    16
#define HD    96
#define QKVN  4608          // 3*NH*HD
#define NT    (NB*TSEQ)     // 4096 rows
#define NTRIP (HD/3)        // 32

typedef __attribute__((ext_vector_type(16))) _Float16 v16h;
typedef __attribute__((ext_vector_type(8)))  _Float16 v8h;
typedef __attribute__((ext_vector_type(8)))  float    v8f;

// ---------------------------------------------------------------------------
// Async global->LDS staging (CDNA5 ASYNCcnt path, probe-confirmed live)
// ---------------------------------------------------------------------------
#if __has_builtin(__builtin_amdgcn_global_load_async_to_lds_b128)
#define HAVE_ASYNC_LDS 1
typedef int v4i __attribute__((vector_size(16)));
typedef __attribute__((address_space(1))) v4i gv4i;   // global int4
typedef __attribute__((address_space(3))) v4i lv4i;   // LDS int4
#else
#define HAVE_ASYNC_LDS 0
#endif

// copy 16 bytes (8 halves) global -> LDS
__device__ __forceinline__ void stage16(const _Float16* g, _Float16* l) {
#if HAVE_ASYNC_LDS
  __builtin_amdgcn_global_load_async_to_lds_b128(
      (gv4i*)(uintptr_t)g, (lv4i*)(unsigned int)(uintptr_t)l, 0, 0);
#else
  *reinterpret_cast<v8h*>(l) = *reinterpret_cast<const v8h*>(g);
#endif
}

__device__ __forceinline__ void wait_async0() {
#if HAVE_ASYNC_LDS
#if __has_builtin(__builtin_amdgcn_s_wait_asynccnt)
  __builtin_amdgcn_s_wait_asynccnt(0);
#else
  asm volatile("s_wait_asynccnt 0" ::: "memory");
#endif
#endif
}

// ---------------------------------------------------------------------------
// 16-lane max reduction with DPP (no LDS / ds_bpermute round-trips).
// DPP rows are 16 lanes, so this stays within each lane-half -> matches the
// WMMA C-layout row groups exactly.
// ---------------------------------------------------------------------------
__device__ __forceinline__ float dpp_max16(float x) {
  union { float f; int i; } u, v;
  u.f = x;
  v.i = __builtin_amdgcn_update_dpp(0, u.i, 0xB1, 0xF, 0xF, true);  // quad_perm [1,0,3,2]
  u.f = fmaxf(u.f, v.f);
  v.i = __builtin_amdgcn_update_dpp(0, u.i, 0x4E, 0xF, 0xF, true);  // quad_perm [2,3,0,1]
  u.f = fmaxf(u.f, v.f);
  v.i = __builtin_amdgcn_update_dpp(0, u.i, 0x141, 0xF, 0xF, true); // row_half_mirror (^7)
  u.f = fmaxf(u.f, v.f);
  v.i = __builtin_amdgcn_update_dpp(0, u.i, 0x140, 0xF, 0xF, true); // row_mirror (^15)
  u.f = fmaxf(u.f, v.f);
  return u.f;
}

// ---------------------------------------------------------------------------
// WMMA helpers — CDNA5 wave32 16x16x32 f16 -> f32
// ---------------------------------------------------------------------------
__device__ __forceinline__ v8f wmma16(v16h a, v16h b, v8f c) {
  // (neg_a, A, neg_b, B, c_mod, C, reuse_a, reuse_b)
  return __builtin_amdgcn_wmma_f32_16x16x32_f16(false, a, false, b, (short)0, c,
                                                false, false);
}

// A-matrix 16x32 fragment: lane = row (m = lane&15); per lane one
// contiguous 16-element K chunk: lanes 0-15 K=0..15, lanes 16-31 K=16..31.
__device__ __forceinline__ v16h load_frag_a(const _Float16* base, int ld, int lane) {
  const int m  = lane & 15;
  const int kb = (lane & 16) ? 16 : 0;
  return *reinterpret_cast<const v16h*>(base + m * ld + kb);
}

// B-matrix 32x16 fragment: lane = column (n = lane&15); per lane one
// contiguous 16-element K chunk: lanes 0-15 K=0..15, lanes 16-31 K=16..31.
__device__ __forceinline__ v16h load_frag_b(const _Float16* base, int ld, int lane) {
  const int n  = lane & 15;
  const int kb = (lane & 16) ? 16 : 0;
  return *reinterpret_cast<const v16h*>(base + n * ld + kb);
}

// ---------------------------------------------------------------------------
// Elementwise cast f32 -> f16
// ---------------------------------------------------------------------------
__global__ void cast_f32_f16(const float* __restrict__ in, _Float16* __restrict__ out,
                             int n) {
  int i = blockIdx.x * blockDim.x + threadIdx.x;
  if (i < n) out[i] = (_Float16)in[i];
}

// ---------------------------------------------------------------------------
// GEMM: C[M,N] = A[M,K] * W[N,K]^T  (f16 in, f32 accumulate).
// One wave computes a 32x64 output tile: 2 A fragments x 4 B fragments.
// ---------------------------------------------------------------------------
template <bool OUT_F16>
__global__ __launch_bounds__(256)
void gemm_nt(const _Float16* __restrict__ A, const _Float16* __restrict__ W,
             void* __restrict__ Cout, int M, int N, int Kd) {
  const int lane = threadIdx.x & 31;
  const int wave = threadIdx.x >> 5;
  const int ntn  = N >> 6;                       // 64-wide col tiles
  const int wid  = blockIdx.x * 8 + wave;
  const int tm   = wid / ntn;
  const int tn   = wid - tm * ntn;
  if (tm * 32 >= M) return;                      // wave-uniform
  const int m0 = tm * 32, n0 = tn * 64;

  v8f z = {};
  v8f acc[2][4];
#pragma unroll
  for (int r = 0; r < 2; ++r)
#pragma unroll
    for (int i = 0; i < 4; ++i) acc[r][i] = z;

  for (int k = 0; k < Kd; k += 32) {
    if (k + 32 < Kd) {  // prefetch next K-slice (global_prefetch_b8)
      __builtin_prefetch(A + (size_t)m0 * Kd + k + 32, 0, 3);
      __builtin_prefetch(W + (size_t)n0 * Kd + k + 32, 0, 3);
    }
    v16h a0 = load_frag_a(A + (size_t)m0 * Kd + k,        Kd, lane);
    v16h a1 = load_frag_a(A + (size_t)(m0 + 16) * Kd + k, Kd, lane);
#pragma unroll
    for (int i = 0; i < 4; ++i) {
      v16h bf = load_frag_b(W + (size_t)(n0 + i * 16) * Kd + k, Kd, lane);
      acc[0][i] = wmma16(a0, bf, acc[0][i]);
      acc[1][i] = wmma16(a1, bf, acc[1][i]);
    }
  }

  const int r0 = (lane & 16) ? 8 : 0;
  const int nn = lane & 15;
#pragma unroll
  for (int r = 0; r < 2; ++r)
#pragma unroll
    for (int i = 0; i < 4; ++i)
#pragma unroll
      for (int j = 0; j < 8; ++j) {
        const size_t idx = (size_t)(m0 + r * 16 + r0 + j) * N + (n0 + i * 16 + nn);
        if (OUT_F16) ((_Float16*)Cout)[idx] = (_Float16)acc[r][i][j];
        else         ((float*)Cout)[idx]    = acc[r][i][j];
      }
}

// ---------------------------------------------------------------------------
// RoPE-3D + QKV split.  Q,K -> (B,H,T,96) f16 (score scale folded into Q).
// V -> TRANSPOSED (B,H,96,T) f16 so attention PV B-fragments are contiguous.
// ---------------------------------------------------------------------------
__global__ void rope_split(const _Float16* __restrict__ qkvh,
                           const float* __restrict__ Rs,
                           _Float16* __restrict__ qh, _Float16* __restrict__ kh,
                           _Float16* __restrict__ vth) {
  const int i = blockIdx.x * blockDim.x + threadIdx.x;
  if (i >= NB * TSEQ * NH * NTRIP) return;
  const int trip =  i        & 31;
  const int h    = (i >> 5)  & 15;
  const int t    = (i >> 9)  & 2047;
  const int b    =  i >> 20;

  const _Float16* row = qkvh + (size_t)(b * TSEQ + t) * QKVN;
  const float* R = Rs + ((size_t)t * NTRIP + trip) * 9;
  const float r00 = R[0], r01 = R[1], r02 = R[2];
  const float r10 = R[3], r11 = R[4], r12 = R[5];
  const float r20 = R[6], r21 = R[7], r22 = R[8];

  const int d  = trip * 3;
  const int cq = h * HD + d;
  const size_t ob = ((size_t)(b * NH + h) * TSEQ + t) * HD + d;
  const float SC = 0.10206207261596575f;  // 1/sqrt(96)

  float x0 = (float)row[cq], x1 = (float)row[cq + 1], x2 = (float)row[cq + 2];
  qh[ob]     = (_Float16)((r00 * x0 + r01 * x1 + r02 * x2) * SC);
  qh[ob + 1] = (_Float16)((r10 * x0 + r11 * x1 + r12 * x2) * SC);
  qh[ob + 2] = (_Float16)((r20 * x0 + r21 * x1 + r22 * x2) * SC);

  x0 = (float)row[DM + cq]; x1 = (float)row[DM + cq + 1]; x2 = (float)row[DM + cq + 2];
  kh[ob]     = (_Float16)(r00 * x0 + r01 * x1 + r02 * x2);
  kh[ob + 1] = (_Float16)(r10 * x0 + r11 * x1 + r12 * x2);
  kh[ob + 2] = (_Float16)(r20 * x0 + r21 * x1 + r22 * x2);

  const size_t vb = ((size_t)(b * NH + h) * HD + d) * TSEQ + t;
  vth[vb]            = row[2 * DM + cq];
  vth[vb + TSEQ]     = row[2 * DM + cq + 1];
  vth[vb + 2 * TSEQ] = row[2 * DM + cq + 2];
}

// ---------------------------------------------------------------------------
// Flash attention (causal, online softmax).  Block = 8 waves = 128 query rows
// of one (b,h); each wave owns a 16-row Q tile.  K/Vt tiles staged via async
// global->LDS DMA.  Row-sum l tracked by a 7th WMMA accumulator (P x ones),
// row-max via DPP butterflies, causal compare/cndmask only on diagonal blocks.
// ---------------------------------------------------------------------------
__global__ __launch_bounds__(256)
void flash_attn(const _Float16* __restrict__ qh, const _Float16* __restrict__ kh,
                const _Float16* __restrict__ vth, _Float16* __restrict__ atth) {
  __shared__ __align__(32) _Float16 lk [32 * HD];      // (key, feat)
  __shared__ __align__(32) _Float16 lvt[HD * 32];      // (feat, key)
  __shared__ __align__(32) _Float16 lp [8 * 16 * 32];  // per-wave P tile

  const int bh   = blockIdx.x;               // 0..31
  const int qblk = blockIdx.y;               // 0..15
  const int b = bh / NH, h = bh % NH;
  const int tid  = threadIdx.x;
  const int wave = tid >> 5;
  const int lane = tid & 31;
  const int q0   = qblk * 128 + wave * 16;

  const _Float16* Q  = qh  + (size_t)(b * NH + h) * TSEQ * HD;
  const _Float16* K  = kh  + (size_t)(b * NH + h) * TSEQ * HD;
  const _Float16* Vt = vth + (size_t)(b * NH + h) * HD * TSEQ;  // (feat, t)

  v16h qf[3];
#pragma unroll
  for (int c = 0; c < 3; ++c)
    qf[c] = load_frag_a(Q + (size_t)q0 * HD + c * 32, HD, lane);

  // all-ones B fragment: P x ones accumulates the flash row-sum "l" in o[6]
  v16h ones;
#pragma unroll
  for (int e = 0; e < 16; ++e) ones[e] = (_Float16)1.0f;

  v8f z = {};
  v8f o[7];
#pragma unroll
  for (int c = 0; c < 7; ++c) o[c] = z;
  float m[8];
#pragma unroll
  for (int j = 0; j < 8; ++j) m[j] = -3.0e38f;

  const int r0 = (lane & 16) ? 8 : 0;
  const int nn = lane & 15;
  _Float16* pb = lp + wave * (16 * 32);

  const int kend = qblk * 128 + 128;         // causal bound for this block
  for (int kb = 0; kb < kend; kb += 32) {
    __syncthreads();
    // stage K tile: 32 rows x 96 feats = 384 x 16B (async DMA to LDS)
    for (int idx = tid; idx < 384; idx += 256) {
      const int r = idx / 12, c = idx % 12;
      stage16(&K[(size_t)(kb + r) * HD + c * 8], &lk[r * HD + c * 8]);
    }
    // stage Vt tile: 96 rows x 32 keys = 384 x 16B (already feat-major)
    for (int idx = tid; idx < 384; idx += 256) {
      const int f = idx >> 2, c = idx & 3;
      stage16(&Vt[(size_t)f * TSEQ + kb + c * 8], &lvt[f * 32 + c * 8]);
    }
    wait_async0();
    __syncthreads();

    // S = Q * K^T  (two 16x16 score tiles: keys kb..kb+15 and kb+16..kb+31)
    v8f s0 = z, s1 = z;
#pragma unroll
    for (int c = 0; c < 3; ++c) {
      s0 = wmma16(qf[c], load_frag_b(&lk[c * 32],           HD, lane), s0);
      s1 = wmma16(qf[c], load_frag_b(&lk[16 * HD + c * 32], HD, lane), s1);
    }

    // causal mask: only the diagonal k-block needs per-element compares
    // (wave-uniform test: keys kb..kb+31 vs this wave's min query row q0)
    float a0[8], a1[8];
    if (kb + 31 > q0) {
      const int col = kb + nn;
#pragma unroll
      for (int j = 0; j < 8; ++j) {
        const int row = q0 + r0 + j;
        a0[j] = (col      <= row) ? s0[j] : -3.0e38f;
        a1[j] = (col + 16 <= row) ? s1[j] : -3.0e38f;
      }
    } else {
#pragma unroll
      for (int j = 0; j < 8; ++j) { a0[j] = s0[j]; a1[j] = s1[j]; }
    }

    // online softmax (row-max via DPP butterfly over 16 lanes)
    float p0[8], p1[8], fc[8];
#pragma unroll
    for (int j = 0; j < 8; ++j) {
      const float rm = dpp_max16(fmaxf(a0[j], a1[j]));
      const float mn = fmaxf(m[j], rm);
      fc[j] = __expf(m[j] - mn);
      p0[j] = __expf(a0[j] - mn);
      p1[j] = __expf(a1[j] - mn);
      m[j] = mn;
    }

    // P (16x32) -> per-wave LDS in row-major, then reload as an A-fragment
#pragma unroll
    for (int j = 0; j < 8; ++j) {
      pb[(r0 + j) * 32 + nn]      = (_Float16)p0[j];
      pb[(r0 + j) * 32 + 16 + nn] = (_Float16)p1[j];
    }
    asm volatile("s_wait_dscnt 0" ::: "memory");
    const v16h pf = load_frag_a(pb, 32, lane);

    // rescale running O (and l in o[6]), accumulate P*V / P*1 via WMMA C
#pragma unroll
    for (int c = 0; c < 7; ++c) {
#pragma unroll
      for (int j = 0; j < 8; ++j) o[c][j] *= fc[j];
    }
#pragma unroll
    for (int c = 0; c < 6; ++c)
      o[c] = wmma16(pf, load_frag_b(&lvt[(c * 16) * 32], 32, lane), o[c]);
    o[6] = wmma16(pf, ones, o[6]);
  }

  // normalize (l = o[6], replicated across the 16 lanes of each half)
  float inv[8];
#pragma unroll
  for (int j = 0; j < 8; ++j) inv[j] = 1.0f / o[6][j];
#pragma unroll
  for (int c = 0; c < 6; ++c)
#pragma unroll
    for (int j = 0; j < 8; ++j) {
      const int t = q0 + r0 + j;
      atth[(size_t)(b * TSEQ + t) * DM + h * HD + c * 16 + nn] =
          (_Float16)(o[c][j] * inv[j]);
    }
}

// ---------------------------------------------------------------------------
// Host-side orchestration
// ---------------------------------------------------------------------------
extern "C" void kernel_launch(void* const* d_in, const int* in_sizes, int n_in,
                              void* d_out, int out_size, void* d_ws, size_t ws_size,
                              hipStream_t stream) {
  const float* x    = (const float*)d_in[0];
  const float* wqkv = (const float*)d_in[1];
  const float* wo   = (const float*)d_in[2];
  const float* Rs   = (const float*)d_in[3];
  float* out = (float*)d_out;
  char*  ws  = (char*)d_ws;

  constexpr size_t SZ_X    = (size_t)NT * DM;       // elements
  constexpr size_t SZ_WQ   = (size_t)QKVN * DM;
  constexpr size_t SZ_WO   = (size_t)DM * DM;
  constexpr size_t SZ_QKV  = (size_t)NT * QKVN;
  constexpr size_t SZ_HEAD = (size_t)NB * NH * TSEQ * HD;  // == SZ_X

  _Float16* xh   = (_Float16*)(ws + 0);
  _Float16* wqh  = (_Float16*)(ws + SZ_X * 2);
  _Float16* woh  = (_Float16*)(ws + SZ_X * 2 + SZ_WQ * 2);
  _Float16* qkvh = (_Float16*)(ws + SZ_X * 2 + SZ_WQ * 2 + SZ_WO * 2);
  _Float16* vth  = (_Float16*)(ws + SZ_X * 2 + SZ_WQ * 2 + SZ_WO * 2 + SZ_QKV * 2);
  _Float16* atth = (_Float16*)(ws + SZ_X * 2 + SZ_WQ * 2 + SZ_WO * 2 + SZ_QKV * 2 +
                               SZ_HEAD * 2);
  _Float16* qh = xh;    // alias (xh dead after QKV GEMM)
  _Float16* kh = wqh;   // alias (wqh dead after QKV GEMM)

  // 1) cast inputs to f16
  cast_f32_f16<<<(int)((SZ_X  + 255) / 256), 256, 0, stream>>>(x,    xh,  (int)SZ_X);
  cast_f32_f16<<<(int)((SZ_WQ + 255) / 256), 256, 0, stream>>>(wqkv, wqh, (int)SZ_WQ);
  cast_f32_f16<<<(int)((SZ_WO + 255) / 256), 256, 0, stream>>>(wo,   woh, (int)SZ_WO);

  // 2) QKV projection: (4096x1536) * (4608x1536)^T -> f16 (4096x4608)
  {
    const int tiles = (NT / 32) * (QKVN / 64);  // 9216 wave-tiles
    gemm_nt<true><<<tiles / 8, 256, 0, stream>>>(xh, wqh, (void*)qkvh, NT, QKVN, DM);
  }

  // 3) RoPE-3D on Q,K (+ fold score scale into Q); V written transposed
  rope_split<<<(NB * TSEQ * NH * NTRIP) / 256, 256, 0, stream>>>(qkvh, Rs, qh, kh, vth);

  // 4) causal flash attention -> atth (B,T,H*96) f16
  flash_attn<<<dim3(NB * NH, TSEQ / 128), 256, 0, stream>>>(qh, kh, vth, atth);

  // 5) output projection: (4096x1536) * (1536x1536)^T -> f32 d_out
  {
    const int tiles = (NT / 32) * (DM / 64);    // 3072 wave-tiles
    gemm_nt<false><<<tiles / 8, 256, 0, stream>>>(atth, woh, (void*)out, NT, DM, DM);
  }
}
